// Quant_Conv2d_34522947125658
// MI455X (gfx1250) — 8-bit matrix operand order repaired and checked by reading, compiled, not run on hardware
//
#include <hip/hip_runtime.h>
#include <stdint.h>

typedef __attribute__((ext_vector_type(8))) int v8i;

#define N_    64
#define CIN   256
#define COUT  256
#define H_    28
#define W_    28
#define HW    784
#define M_TOTAL (N_ * HW)
#define KTAPS 9
#define KDIM  (CIN * KTAPS)
#define OUTQ_ELEMS (N_ * COUT * HW)
#define BSTRIDE 272

__global__ void pack_x_kernel(const int* __restrict__ xq,
                              unsigned char* __restrict__ xu8) {
    int i = blockIdx.x * blockDim.x + threadIdx.x;
    if (i >= N_ * HW * CIN) return;
    int c   = i & (CIN - 1);
    int pix = i >> 8;
    int w = pix % W_;
    int t = pix / W_;
    int h = t % H_;
    int n = t / H_;
    int src = ((n * CIN + c) * H_ + h) * W_ + w;
    xu8[i] = (unsigned char)xq[src];
}

__global__ void quant_w_kernel(const float* __restrict__ weight,
                               const float* __restrict__ scale_x,
                               const float* __restrict__ bias,
                               signed char* __restrict__ wq,
                               int* __restrict__ bq,
                               float* __restrict__ scale_out) {
    __shared__ float red[256];
    const int co  = blockIdx.x;
    const int tid = threadIdx.x;
    const float* wrow = weight + (size_t)co * KDIM;

    float amax = 0.0f;
    for (int i = tid; i < KDIM; i += 256) amax = fmaxf(amax, fabsf(wrow[i]));
    red[tid] = amax;
    __syncthreads();
    for (int s = 128; s > 0; s >>= 1) {
        if (tid < s) red[tid] = fmaxf(red[tid], red[tid + s]);
        __syncthreads();
    }
    const float scale_w = fmaxf(red[0], 1e-8f) / 127.0f;

    for (int i = tid; i < KDIM; i += 256) {
        int ci  = i / KTAPS;
        int tap = i % KTAPS;
        float q = rintf(wrow[i] / scale_w);
        q = fminf(fmaxf(q, -128.0f), 127.0f);
        wq[(size_t)co * KDIM + tap * CIN + ci] = (signed char)q;
    }

    if (tid == 0) {
        float so = scale_w * scale_x[0];
        scale_out[co] = so;
        double r = rint((double)bias[co] / (double)so);
        r = fmin(fmax(r, -2147483648.0), 2147483647.0);
        bq[co] = (int)(long long)r;
    }
}

__device__ __forceinline__ void async_load_b128(unsigned lds_off,
                                                const void* gaddr) {
    asm volatile("global_load_async_to_lds_b128 %0, %1, off"
                 :: "v"(lds_off),
                    "v"((unsigned long long)(uintptr_t)gaddr)
                 : "memory");
}
__device__ __forceinline__ void wait_asynccnt0() {
    asm volatile("s_wait_asynccnt 0x0" ::: "memory");
}

__global__ __launch_bounds__(64) void conv_wmma_kernel(
        const unsigned char* __restrict__ xu8,
        const signed char*  __restrict__ wq,
        const int*          __restrict__ bq,
        int*                __restrict__ outq) {
    __shared__ __align__(16) signed char bsm[2][64 * BSTRIDE];

    const int tid    = threadIdx.x;
    const int lane   = tid & 31;
    const int wave   = tid >> 5;
    const int m_base  = blockIdx.x * 128 + wave * 64;
    const int co_base = blockIdx.y * 64;
    const int hi      = (lane >= 16) ? 1 : 0;
    const int l16     = lane & 15;

    const unsigned lds_base = (unsigned)(uintptr_t)(void*)&bsm[0][0];

    const signed char* fsrc0 = wq + (size_t)(co_base + tid) * KDIM;
    const unsigned     fdst0 = lds_base + (unsigned)(tid * BSTRIDE);
    auto fill_tap = [&](int buf, int tap) {
        const signed char* src = fsrc0 + tap * CIN;
        unsigned dst = fdst0 + (unsigned)(buf * (64 * BSTRIDE));
#pragma unroll
        for (int j = 0; j < 16; ++j)
            async_load_b128(dst + j * 16, src + j * 16);
    };

    int rowN[4], rowH[4], rowW[4];
#pragma unroll
    for (int mi = 0; mi < 4; ++mi) {
        int m = m_base + mi * 16 + l16;
        int w = m % W_;
        int t = m / W_;
        rowW[mi] = w;
        rowH[mi] = t % H_;
        rowN[mi] = t / H_;
    }

    unsigned boff[4];
#pragma unroll
    for (int ni = 0; ni < 4; ++ni)
        boff[ni] = (unsigned)((ni * 16 + l16) * BSTRIDE + hi * 16);

    v8i acc[4][4];
#pragma unroll
    for (int mi = 0; mi < 4; ++mi)
#pragma unroll
        for (int ni = 0; ni < 4; ++ni)
            acc[mi][ni] = (v8i)0;

    fill_tap(0, 0);
    wait_asynccnt0();
    __syncthreads();

    for (int tap = 0; tap < KTAPS; ++tap) {
        const int kh  = tap / 3;
        const int kw  = tap % 3;
        const int cur = tap & 1;

        if (tap + 1 < KTAPS) fill_tap(cur ^ 1, tap + 1);

        const unsigned char* abase[4];
        bool avalid[4];
#pragma unroll
        for (int mi = 0; mi < 4; ++mi) {
            int ih = rowH[mi] + kh - 1;
            int iw = rowW[mi] + kw - 1;
            avalid[mi] = ((unsigned)ih < (unsigned)H_) &&
                         ((unsigned)iw < (unsigned)W_);
            abase[mi]  = xu8 +
                ((size_t)(rowN[mi] * HW + ih * W_ + iw)) * CIN + hi * 16;
            if (avalid[mi]) __builtin_prefetch(abase[mi], 0, 0);
        }

        const signed char* bbuf = &bsm[cur][0];

#pragma unroll
        for (int ks = 0; ks < 4; ++ks) {
            const int k0 = ks * 64;

            v8i afrag[4];
#pragma unroll
            for (int mi = 0; mi < 4; ++mi) {
                v8i a = (v8i)0;
                if (avalid[mi]) {
                    const unsigned char* p = abase[mi] + k0;
                    uint2 d0 = *(const uint2*)(p);
                    uint2 d1 = *(const uint2*)(p + 8);
                    uint2 d2 = *(const uint2*)(p + 32);
                    uint2 d3 = *(const uint2*)(p + 40);
                    a[0] = (int)d0.x; a[1] = (int)d0.y;
                    a[2] = (int)d1.x; a[3] = (int)d1.y;
                    a[4] = (int)d2.x; a[5] = (int)d2.y;
                    a[6] = (int)d3.x; a[7] = (int)d3.y;
                }
                afrag[mi] = a;
            }

#pragma unroll
            for (int ni = 0; ni < 4; ++ni) {
                const signed char* p = bbuf + boff[ni] + k0;
                int4 q0 = *(const int4*)(p);
                int4 q1 = *(const int4*)(p + 32);
                v8i b;
                b[0] = q0.x; b[1] = q0.y; b[2] = q0.z; b[3] = q0.w;
                b[4] = q1.x; b[5] = q1.y; b[6] = q1.z; b[7] = q1.w;

#pragma unroll
                for (int mi = 0; mi < 4; ++mi)
                    acc[mi][ni] = __builtin_amdgcn_wmma_i32_16x16x64_iu8(
                         false, afrag[mi],
                         true,  b,
                        acc[mi][ni],
                         false,  false);
            }
        }

        if (tap + 1 < KTAPS) {
            wait_asynccnt0();
            __syncthreads();
        }
    }

#pragma unroll
    for (int ni = 0; ni < 4; ++ni) {
        int co = co_base + ni * 16 + l16;
        int bias_q   = bq[co];
        size_t cobase = (size_t)co * HW;
#pragma unroll
        for (int mi = 0; mi < 4; ++mi) {
#pragma unroll
            for (int r = 0; r < 8; ++r) {
                int m = m_base + mi * 16 + r + hi * 8;
                int n = m / HW;
                size_t idx = cobase + (size_t)m + (size_t)n * (HW * (COUT - 1));
                outq[idx] = acc[mi][ni][r] + bias_q;
            }
        }
    }
}

extern "C" void kernel_launch(void* const* d_in, const int* in_sizes, int n_in,
                              void* d_out, int out_size, void* d_ws, size_t ws_size,
                              hipStream_t stream) {
    const int*   xq      = (const int*)d_in[0];
    const float* scale_x = (const float*)d_in[1];
    const float* weight  = (const float*)d_in[2];
    const float* bias    = (const float*)d_in[3];

    int*   outq      = (int*)d_out;
    float* scale_out = (float*)d_out + OUTQ_ELEMS;

    unsigned char* xu8 = (unsigned char*)d_ws;
    signed char*   wqp = (signed char*)d_ws + OUTQ_ELEMS;
    int*           bqp = (int*)((char*)d_ws + OUTQ_ELEMS + (size_t)COUT * KDIM);

    pack_x_kernel<<<(N_ * HW * CIN + 255) / 256, 256, 0, stream>>>(xq, xu8);
    quant_w_kernel<<<COUT, 256, 0, stream>>>(weight, scale_x, bias,
                                             wqp, bqp, scale_out);
    dim3 grid(M_TOTAL / 128, COUT / 64);
    conv_wmma_kernel<<<grid, 64, 0, stream>>>(xu8, wqp, bqp, outq);
}
